// Network_clip_add_dgcnn_acc_cvae_24713241821968
// MI455X (gfx1250) — matrix operand order repaired and checked by reading, not compiled, not run on hardware
//
#include <hip/hip_runtime.h>
#include <hip/hip_bf16.h>

// ---------------------------------------------------------------------------
// Shapes fixed by the reference: B=2, N=M=512, D=128, H=256.
// ---------------------------------------------------------------------------
#define BATCH 2
#define NPTS  512
#define DEMB  128
#define HIDW  256

typedef __attribute__((ext_vector_type(16))) __bf16 v16bf;
typedef __attribute__((ext_vector_type(8)))  float  v8f;

// ---------------------------------------------------------------------------
// Kernel 1: embedding linears.
//   hA[b,n,h] = sum_d Phi_A[b,n,d] * W1[d,h]        (d in [0,128))
//   hB[b,m,h] = sum_d Phi_B[b,m,d] * W1[128+d,h]
// grid = (B*N, 2), block = 256 (one thread per h).
// ---------------------------------------------------------------------------
__global__ __launch_bounds__(256) void emb_linear_kernel(
    const float* __restrict__ PhiA, const float* __restrict__ PhiB,
    const float* __restrict__ W1,
    float* __restrict__ hA, float* __restrict__ hB) {
  __shared__ float sphi[DEMB];
  const int row  = blockIdx.x;           // b*N + n
  const int side = blockIdx.y;           // 0 -> A, 1 -> B
  const int tid  = threadIdx.x;          // h

  const float* phi = (side == 0 ? PhiA : PhiB) + (size_t)row * DEMB;
  if (tid < DEMB) sphi[tid] = phi[tid];
  __syncthreads();

  const int woff = side * DEMB;          // row offset into W1 [256 x 256]
  float acc = 0.f;
#pragma unroll 8
  for (int d = 0; d < DEMB; ++d)
    acc = fmaf(sphi[d], W1[(size_t)(d + woff) * HIDW + tid], acc);

  float* dst = (side == 0 ? hA : hB);
  dst[(size_t)row * HIDW + tid] = acc;
}

// ---------------------------------------------------------------------------
// Kernel 2: pack W2 [256 x 256] (row-major, k-major) into bf16 WMMA
// B-fragment order. One fragment = (col-tile j, k-step ks): B tile is
// 32(K) x 16(N).  ISA B layout (16-bit): lane l holds col n = l&15,
// K = 16*(e>>3) + 8*(l>>4) + (e&7) for vector element e (0..15).
// Storage: w2f[(((j*8 + ks)*32 + lane)*16 + e]  -> 32B contiguous per lane.
// ---------------------------------------------------------------------------
__global__ __launch_bounds__(256) void w2_pack_kernel(
    const float* __restrict__ W2, __bf16* __restrict__ w2f) {
  const int t = blockIdx.x * 256 + threadIdx.x;    // 0 .. 65535
  const int e    = t & 15;
  const int lane = (t >> 4) & 31;
  const int ksj  = t >> 9;                          // j*8 + ks
  const int ks   = ksj & 7;
  const int j    = ksj >> 3;
  const int half = lane >> 4;
  const int n    = lane & 15;
  const int k    = ks * 32 + 16 * (e >> 3) + 8 * half + (e & 7); // W2 row (K)
  const int c    = j * 16 + n;                      // W2 col (H out)
  w2f[t] = (__bf16)W2[(size_t)k * HIDW + c];
}

// ---------------------------------------------------------------------------
// Kernel 3: per-pair MLP via WMMA.
// grid = B * (N/16) * (N/16) blocks, 512 threads = 16 wave32s.
// Block owns pair tile (n0..n0+15) x (m0..m0+15).  Wave w fixes n = n0+w;
// its 16 X-rows are m_local = 0..15.
//   A frag (16-bit, 16x32): lane holds row M = lane&15 (= m_local),
//     element e holds K = 16*(e>>3) + 8*(lane>>4) + (e&7).
//   C/D (f32 16x16): vgpr r, lane l -> row r + 8*(l>>4), col l&15.
// WMMA chains start from inline C=0 (no init movs / no WAW hazard nops);
// b2 + relu + W3-dot are fused in the register epilogue.
// ---------------------------------------------------------------------------
__global__ __launch_bounds__(512) void pair_mlp_wmma_kernel(
    const float* __restrict__ hA, const float* __restrict__ hB,
    const float* __restrict__ b1, const __bf16* __restrict__ w2f,
    const float* __restrict__ b2, const float* __restrict__ W3,
    const float* __restrict__ b3p, float* __restrict__ out) {
  __shared__ float sA[16][HIDW];   // hA rows n0..n0+15, b1 pre-added
  __shared__ float sB[16][HIDW];   // hB rows m0..m0+15
  __shared__ float sb2[HIDW];
  __shared__ float sw3[HIDW];

  const int tiles = NPTS / 16;                       // 32
  const int blk = blockIdx.x;
  const int b   = blk / (tiles * tiles);
  const int rem = blk % (tiles * tiles);
  const int n0  = (rem / tiles) * 16;
  const int m0  = (rem % tiles) * 16;

  const int tid = threadIdx.x;
  for (int i = tid; i < 16 * HIDW; i += 512) {
    const int r = i >> 8, h = i & (HIDW - 1);
    sA[r][h] = hA[((size_t)b * NPTS + n0 + r) * HIDW + h] + b1[h];
    sB[r][h] = hB[((size_t)b * NPTS + m0 + r) * HIDW + h];
  }
  for (int h = tid; h < HIDW; h += 512) { sb2[h] = b2[h]; sw3[h] = W3[h]; }
  __syncthreads();

  const int wave = tid >> 5;        // 0..15  -> n = n0 + wave
  const int lane = tid & 31;
  const int half = lane >> 4;
  const int col  = lane & 15;       // A-row (m_local) AND D-col index

  // Build the 8 A fragments (K = 0..255) once; reused by all 16 col-tiles.
  v16bf a[8];
#pragma unroll
  for (int ks = 0; ks < 8; ++ks) {
#pragma unroll
    for (int e = 0; e < 16; ++e) {
      const int K = 16 * (e >> 3) + 8 * half + (e & 7);
      const int h = ks * 32 + K;
      const float v = sA[wave][h] + sB[col][h];       // hA + hB + b1
      a[ks][e] = (__bf16)fmaxf(v, 0.f);               // relu(layer1)
    }
  }

  // Hoist per-lane b2 / W3 values for all 16 col-tiles out of the hot loop.
  float b2v[16], w3v[16];
#pragma unroll
  for (int j = 0; j < 16; ++j) {
    b2v[j] = sb2[j * 16 + col];
    w3v[j] = sw3[j * 16 + col];
  }

  float pr[8];
#pragma unroll
  for (int r = 0; r < 8; ++r) pr[r] = 0.f;

#pragma unroll
  for (int j = 0; j < 16; ++j) {                      // output col-tiles
    v8f acc = {};                                     // -> inline C = 0

#pragma unroll
    for (int ks = 0; ks < 8; ++ks) {
      const v16bf bf =
          *(const v16bf*)(w2f + (size_t)(((j * 8 + ks) * 32 + lane) * 16));
      acc = __builtin_amdgcn_wmma_f32_16x16x32_bf16(
          false, a[ks], false, bf, (short)0, acc, false, false);
    }

#pragma unroll
    for (int r = 0; r < 8; ++r) {
      const float y = fmaxf(acc[r] + b2v[j], 0.f);    // +b2, relu(layer2)
      pr[r] = fmaf(y, w3v[j], pr[r]);                 // fused W3 dot
    }
  }

  // Sum over the 16 D-columns (lanes within each half-group).
#pragma unroll
  for (int r = 0; r < 8; ++r) {
    float v = pr[r];
    v += __shfl_xor(v, 1, 32);
    v += __shfl_xor(v, 2, 32);
    v += __shfl_xor(v, 4, 32);
    v += __shfl_xor(v, 8, 32);
    pr[r] = v;
  }

  const float b3 = b3p[0];
  const size_t obase = ((size_t)b * NPTS + (n0 + wave)) * NPTS;
  const int mbase = m0 + 8 * half;                    // rows 0-7 / 8-15
#pragma unroll
  for (int r = 0; r < 8; ++r)
    if (col == r) out[obase + mbase + r] = pr[r] + b3;
}

// ---------------------------------------------------------------------------
// Launcher. Inputs (all f32): Phi_A, Phi_B, W1, b1, W2, b2, W3, b3.
// Workspace: hA (1MB) | hB (1MB) | w2f bf16 (128KB).
// ---------------------------------------------------------------------------
extern "C" void kernel_launch(void* const* d_in, const int* in_sizes, int n_in,
                              void* d_out, int out_size, void* d_ws, size_t ws_size,
                              hipStream_t stream) {
  (void)in_sizes; (void)n_in; (void)out_size; (void)ws_size;
  const float* PhiA = (const float*)d_in[0];
  const float* PhiB = (const float*)d_in[1];
  const float* W1   = (const float*)d_in[2];
  const float* b1   = (const float*)d_in[3];
  const float* W2   = (const float*)d_in[4];
  const float* b2   = (const float*)d_in[5];
  const float* W3   = (const float*)d_in[6];
  const float* b3   = (const float*)d_in[7];
  float* out = (float*)d_out;

  char* ws = (char*)d_ws;
  float*  hA  = (float*)(ws);
  float*  hB  = (float*)(ws + (size_t)BATCH * NPTS * HIDW * sizeof(float));
  __bf16* w2f = (__bf16*)(ws + 2 * (size_t)BATCH * NPTS * HIDW * sizeof(float));

  // 1) embeddings: hA / hB
  emb_linear_kernel<<<dim3(BATCH * NPTS, 2), 256, 0, stream>>>(
      PhiA, PhiB, W1, hA, hB);

  // 2) pack W2 into bf16 WMMA B-fragments
  w2_pack_kernel<<<(HIDW * HIDW) / 256, 256, 0, stream>>>(W2, w2f);

  // 3) pair MLP with WMMA
  const int nblocks = BATCH * (NPTS / 16) * (NPTS / 16);  // 2048
  pair_mlp_wmma_kernel<<<nblocks, 512, 0, stream>>>(
      hA, hB, b1, w2f, b2, W3, b3, out);
}
